// GLCA_25005299598027
// MI455X (gfx1250) — compile-verified, matrix operand order checked by reading, not run on hardware
//
#include <hip/hip_runtime.h>
#include <hip/hip_bf16.h>

#define DEVINL __device__ __forceinline__

typedef __attribute__((ext_vector_type(16))) __bf16          v16bf;
typedef __attribute__((ext_vector_type(16))) unsigned short  us16;
typedef __attribute__((ext_vector_type(8)))  unsigned short  us8;
typedef __attribute__((ext_vector_type(8)))  float           v8f;

constexpr int Bn = 8, Cc = 256, Hh = 160, Ww = 160, ICn = 32;
constexpr int HW = Hh * Ww;        // 25600
constexpr int P  = Bn * HW;        // 204800 pixels
constexpr int BW = Bn * Ww;        // 1280 attention slices

DEVINL unsigned short f2bf(float f) {
  unsigned int u = __builtin_bit_cast(unsigned int, f);
  u = u + 0x7FFFu + ((u >> 16) & 1u);            // round-to-nearest-even
  return (unsigned short)(u >> 16);
}

DEVINL v8f v8f_zero() {
  v8f v;
#pragma unroll
  for (int i = 0; i < 8; ++i) v[i] = 0.0f;
  return v;
}

DEVINL us16 us16_zero() {
  us16 v;
#pragma unroll
  for (int i = 0; i < 16; ++i) v[i] = 0;
  return v;
}

DEVINL v8f wmma_bf16(us16 a, us16 b, v8f c) {
  return __builtin_amdgcn_wmma_f32_16x16x32_bf16(
      false, __builtin_bit_cast(v16bf, a),
      false, __builtin_bit_cast(v16bf, b),
      (short)0, c, false, false);
}

// ---- CDNA5 async memory->LDS path (ASYNCcnt-tracked DMA-style copy) -------
DEVINL unsigned lds_off(const void* p) { return (unsigned)(size_t)p; }

DEVINL void async_load_b128(unsigned lds_addr, unsigned voff_bytes,
                            unsigned long long base) {
  asm volatile("global_load_async_to_lds_b128 %0, %1, %2"
               :
               : "v"(lds_addr), "v"(voff_bytes), "s"(base)
               : "memory");
}

DEVINL void wait_async() {
  asm volatile("s_wait_asynccnt 0x0" ::: "memory");
}

// ---------------------------------------------------------------------------
// K0: feature [B,C,H,W] f32 -> Xb [P, 256] bf16 (pixel-major). Coalesced f32
// reads, packed 16B bf16 stores via LDS transpose.
// ---------------------------------------------------------------------------
__global__ void k_transpose(const float* __restrict__ feat,
                            unsigned short* __restrict__ Xb) {
  __shared__ float tile[32][65];
  int tid = threadIdx.x;
  int pbase = blockIdx.x * 64;                       // 64 pixels per block
  int b = pbase / HW;
  int off = pbase % HW;
  for (int cb = 0; cb < 8; ++cb) {
#pragma unroll
    for (int i = 0; i < 8; ++i) {                    // 32c x 64p floats
      int idx = tid + i * 256;
      int r = idx >> 6, col = idx & 63;
      tile[r][col] = feat[(size_t)(b * Cc + cb * 32 + r) * HW + off + col];
    }
    __syncthreads();
    {
      int pl = tid >> 2, c0 = (tid & 3) * 8;         // pack 8 channels -> 16B
      us8 v;
#pragma unroll
      for (int j = 0; j < 8; ++j) v[j] = f2bf(tile[c0 + j][pl]);
      *(us8*)&Xb[(size_t)(pbase + pl) * Cc + cb * 32 + c0] = v;
    }
    __syncthreads();
  }
}

// ---------------------------------------------------------------------------
// K_prep: pack weights bf16 + fold BN into per-channel scale/bias
// ---------------------------------------------------------------------------
struct PrepIn {
  const float *qw, *kw, *vw;
  const float *qg, *qb, *qm, *qv;
  const float *kg, *kbt, *km, *kv;
  const float *vg, *vb, *vm, *vv;
  const float *qa, *ka, *va, *oa;
  const float *d1w, *d2w, *d3w;
  const float *dg[3], *db[3], *dm[3], *dv[3];
  const float *ow, *og, *ob, *om, *ov;
};
struct PrepOut {
  unsigned short *wqkv, *wd, *wo;
  float *sqkv, *bqkv, *sd, *bd, *so, *bo, *alphas;
};

__global__ void k_prep(PrepIn in, PrepOut o) {
  int tid = threadIdx.x;
  for (int i = tid; i < 96 * 256; i += 256) {
    int n = i >> 8, c = i & 255;
    const float* src = n < 32 ? in.qw : n < 64 ? in.kw : in.vw;
    o.wqkv[i] = f2bf(src[(n & 31) * 256 + c]);
  }
  for (int i = tid; i < 3 * 9 * 32 * 32; i += 256) {
    int br = i / 9216, r = i % 9216;
    int t = r >> 10, r2 = r & 1023;
    int oc = r2 >> 5, ic = r2 & 31;
    const float* w = br == 0 ? in.d1w : br == 1 ? in.d2w : in.d3w;
    o.wd[i] = f2bf(w[((oc * 32 + ic) * 3 + t / 3) * 3 + (t % 3)]);
  }
  for (int i = tid; i < 256 * 96; i += 256) o.wo[i] = f2bf(in.ow[i]);
  for (int i = tid; i < 96; i += 256) {
    int g = i >> 5, j = i & 31;
    const float* ga = g == 0 ? in.qg : g == 1 ? in.kg : in.vg;
    const float* be = g == 0 ? in.qb : g == 1 ? in.kbt : in.vb;
    const float* me = g == 0 ? in.qm : g == 1 ? in.km : in.vm;
    const float* va = g == 0 ? in.qv : g == 1 ? in.kv : in.vv;
    float s = ga[j] * rsqrtf(va[j] + 1e-5f);
    o.sqkv[i] = s;
    o.bqkv[i] = be[j] - me[j] * s;
    float s2 = in.dg[g][j] * rsqrtf(in.dv[g][j] + 1e-5f);
    o.sd[i] = s2;
    o.bd[i] = in.db[g][j] - in.dm[g][j] * s2;
  }
  for (int i = tid; i < 256; i += 256) {
    float s = in.og[i] * rsqrtf(in.ov[i] + 1e-5f);
    o.so[i] = s;
    o.bo[i] = in.ob[i] - in.om[i] * s;
  }
  if (tid == 0) {
    o.alphas[0] = in.qa[0]; o.alphas[1] = in.ka[0];
    o.alphas[2] = in.va[0]; o.alphas[3] = in.oa[0];
  }
}

// ---------------------------------------------------------------------------
// K1: fused QKV GEMM  Xb[P,256] x Wqkv^T[256,96] + BN + PReLU
//     Async-to-LDS double-buffered pipeline: DMA of tile kb+1 overlaps the
//     6 WMMAs of tile kb; one barrier + one asynccnt wait per k-step.
// ---------------------------------------------------------------------------
__global__ void k_qkv(const unsigned short* __restrict__ Xb,
                      const unsigned short* __restrict__ wqkv,
                      const float* __restrict__ sqkv,
                      const float* __restrict__ bqkv,
                      const float* __restrict__ alphas,
                      unsigned short* __restrict__ q_s,
                      unsigned short* __restrict__ k_s,
                      unsigned short* __restrict__ val_s) {
  __shared__ __align__(32) unsigned short wLDS[96 * 256];     // 48 KB
  __shared__ __align__(32) unsigned short aLDS[2][128 * 32];  // 2 x 8 KB
  int tid = threadIdx.x, lane = tid & 31, wave = tid >> 5;
  int pbase = blockIdx.x * 128;
  unsigned long long wbase = (unsigned long long)(size_t)wqkv;
  unsigned long long xbase = (unsigned long long)(size_t)Xb;
  // weights: 3072 16B chunks, 12 per thread, unconditional
#pragma unroll
  for (int i = 0; i < 12; ++i) {
    int c = tid + i * 256;
    async_load_b128(lds_off(&wLDS[c * 8]), (unsigned)(c * 16), wbase);
  }
  // A tile kb: 512 16B chunks, 2 per thread, unconditional
  int c0 = tid, c1 = tid + 256;
  int r0 = c0 >> 2, o0 = (c0 & 3) * 8;
  int r1 = c1 >> 2, o1 = (c1 & 3) * 8;
  unsigned lds0a = lds_off(&aLDS[0][r0 * 32 + o0]);
  unsigned lds0b = lds_off(&aLDS[0][r1 * 32 + o1]);
  unsigned lds1a = lds_off(&aLDS[1][r0 * 32 + o0]);
  unsigned lds1b = lds_off(&aLDS[1][r1 * 32 + o1]);
  unsigned ga = (unsigned)(((size_t)(pbase + r0) * Cc + o0) * 2);
  unsigned gb = (unsigned)(((size_t)(pbase + r1) * Cc + o1) * 2);

  async_load_b128(lds0a, ga, xbase);
  async_load_b128(lds0b, gb, xbase);
  wait_async();
  __syncthreads();

  v8f acc[6];
#pragma unroll
  for (int nt = 0; nt < 6; ++nt) acc[nt] = v8f_zero();
#pragma unroll
  for (int kb = 0; kb < 8; ++kb) {
    if (kb < 7) {  // prefetch next tile into the other buffer (DMA overlaps WMMA)
      unsigned na = ((kb + 1) & 1) ? lds1a : lds0a;
      unsigned nb = ((kb + 1) & 1) ? lds1b : lds0b;
      async_load_b128(na, ga + (kb + 1) * 64, xbase);
      async_load_b128(nb, gb + (kb + 1) * 64, xbase);
    }
    us16 a = *(const us16*)&aLDS[kb & 1][(wave * 16 + (lane & 15)) * 32 + (lane >> 4) * 16];
#pragma unroll
    for (int nt = 0; nt < 6; ++nt) {
      us16 b = *(const us16*)&wLDS[(nt * 16 + (lane & 15)) * 256 + kb * 32 + (lane >> 4) * 16];
      acc[nt] = wmma_bf16(a, b, acc[nt]);
    }
    wait_async();
    __syncthreads();
  }
#pragma unroll
  for (int nt = 0; nt < 6; ++nt) {
    int n = nt * 16 + (lane & 15);
    float s = sqkv[n], bi = bqkv[n];
    float al = alphas[n >> 5];
#pragma unroll
    for (int r = 0; r < 8; ++r) {
      int m = pbase + wave * 16 + r + ((lane >> 4) << 3);
      float v = acc[nt][r] * s + bi;
      v = v >= 0.f ? v : al * v;
      unsigned short hv = f2bf(v);
      int b = m / HW, off = m % HW;
      int h = off / Ww, w = off % Ww;
      int bw = b * Ww + w;
      int ic = n & 31;
      if (n < 32)       q_s[((size_t)(bw * Hh + h)) * ICn + ic] = hv;
      else if (n < 64)  k_s[((size_t)(bw * Hh + h)) * ICn + ic] = hv;
      else              val_s[((size_t)((b * Hh + h) * Ww + w)) * ICn + ic] = hv;
    }
  }
}

// ---------------------------------------------------------------------------
// K2: dilated 3x3 conv as implicit GEMM (9 taps x K=32), relu(bn(.)) fused.
//     out tcat: [br][BW][IC][H] bf16  == permute(0,3,1,2) layout
// ---------------------------------------------------------------------------
__global__ void k_dconv(const unsigned short* __restrict__ val_s,
                        const unsigned short* __restrict__ wd,
                        const float* __restrict__ sd,
                        const float* __restrict__ bd,
                        unsigned short* __restrict__ tcat) {
  __shared__ __align__(32) unsigned short wLDS[9 * 32 * 32];  // 18 KB
  int tid = threadIdx.x, lane = tid & 31, wave = tid >> 5;
  int br = blockIdx.y;
  int d = (br == 0) ? 1 : (br == 1) ? 3 : 5;
  int pbase = blockIdx.x * 128;
#pragma unroll
  for (int i = 0; i < 4; ++i) {                               // 1024 chunks
    int c = tid + i * 256;
    async_load_b128(lds_off(&wLDS[c * 8]),
                    (unsigned)(br * 18432 + c * 16),
                    (unsigned long long)(size_t)wd);
  }
  if (tid < 128)                                              // remaining 128
    async_load_b128(lds_off(&wLDS[(tid + 1024) * 8]),
                    (unsigned)(br * 18432 + (tid + 1024) * 16),
                    (unsigned long long)(size_t)wd);
  wait_async();
  __syncthreads();
  v8f acc[2];
  acc[0] = v8f_zero(); acc[1] = v8f_zero();
  int m = pbase + wave * 16 + (lane & 15);
  int b = m / HW, off = m % HW;
  int h = off / Ww, w = off % Ww;
  int koff = (lane >> 4) * 16;
#pragma unroll
  for (int t = 0; t < 9; ++t) {
    int hh = h + (t / 3 - 1) * d;
    int ww = w + (t % 3 - 1) * d;
    us16 a = us16_zero();
    if (hh >= 0 && hh < Hh && ww >= 0 && ww < Ww)
      a = *(const us16*)&val_s[((size_t)((b * Hh + hh) * Ww + ww)) * ICn + koff];
    us16 b0 = *(const us16*)&wLDS[(t * 32 + (lane & 15)) * 32 + koff];
    us16 b1 = *(const us16*)&wLDS[(t * 32 + 16 + (lane & 15)) * 32 + koff];
    acc[0] = wmma_bf16(a, b0, acc[0]);
    acc[1] = wmma_bf16(a, b1, acc[1]);
  }
#pragma unroll
  for (int nt = 0; nt < 2; ++nt) {
    int oc = nt * 16 + (lane & 15);
    float s = sd[br * 32 + oc], bi = bd[br * 32 + oc];
#pragma unroll
    for (int r = 0; r < 8; ++r) {
      int mm = pbase + wave * 16 + r + ((lane >> 4) << 3);
      float v = acc[nt][r] * s + bi;
      v = v > 0.f ? v : 0.f;                       // relu
      int bb = mm / HW, o2 = mm % HW;
      int hh2 = o2 / Ww, ww2 = o2 % Ww;
      int bw = bb * Ww + ww2;
      tcat[(((size_t)(br * BW + bw)) * ICn + oc) * Hh + hh2] = f2bf(v);
    }
  }
}

// ---------------------------------------------------------------------------
// K3: per (bw, 16-row tile): S = Q K^T (WMMA) -> softmax -> P @ Tview (WMMA)
//     acat: [br][BW][H][IC] bf16  (branch-major concat)
// ---------------------------------------------------------------------------
__global__ void k_attn(const unsigned short* __restrict__ q_s,
                       const unsigned short* __restrict__ k_s,
                       const unsigned short* __restrict__ tcat,
                       unsigned short* __restrict__ acat) {
  __shared__ __align__(32) unsigned short Qs[16 * 32];       // 1 KB
  __shared__ __align__(32) unsigned short Ks[160 * 32];      // 10 KB
  __shared__ __align__(32) unsigned short Tt[3 * 32 * 160];  // 30 KB
  __shared__ __align__(32) float          Sf[16 * 160];      // 10 KB
  __shared__ __align__(32) unsigned short Pl[16 * 160];      // 5 KB
  int tid = threadIdx.x, lane = tid & 31, wave = tid >> 5;
  int mt = blockIdx.x;     // 0..9 : 16-row tile of att
  int bw = blockIdx.y;     // 0..1279
  if (tid < 64)                                               // Q tile
    async_load_b128(lds_off(&Qs[tid * 8]),
                    (unsigned)(((size_t)(bw * Hh + mt * 16) * ICn) * 2 + tid * 16),
                    (unsigned long long)(size_t)q_s);
#pragma unroll
  for (int i = 0; i < 2; ++i) {                               // K slice: 640 chunks
    int c = tid + i * 256;
    async_load_b128(lds_off(&Ks[c * 8]),
                    (unsigned)(((size_t)bw * Hh * ICn) * 2 + c * 16),
                    (unsigned long long)(size_t)k_s);
  }
  if (tid < 128)
    async_load_b128(lds_off(&Ks[(tid + 512) * 8]),
                    (unsigned)(((size_t)bw * Hh * ICn) * 2 + (tid + 512) * 16),
                    (unsigned long long)(size_t)k_s);
  for (int i = tid; i < 1920; i += 256) {                     // Tview^T gather
    int br = i / 640, rem = i % 640;
    int h2 = rem >> 2, j0 = (rem & 3) * 8;
    us8 v = *(const us8*)&tcat[(size_t)(br * BW + bw) * 5120 + h2 * 32 + j0];
#pragma unroll
    for (int j = 0; j < 8; ++j) Tt[(br * 32 + j0 + j) * 160 + h2] = v[j];
  }
  wait_async();
  __syncthreads();
  // S = Q K^T : one K=32 WMMA per 16x16 tile
  us16 aq = *(const us16*)&Qs[(lane & 15) * 32 + (lane >> 4) * 16];
  for (int nt = wave; nt < 10; nt += 8) {
    us16 b = *(const us16*)&Ks[(nt * 16 + (lane & 15)) * 32 + (lane >> 4) * 16];
    v8f c = v8f_zero();
    c = wmma_bf16(aq, b, c);
#pragma unroll
    for (int r = 0; r < 8; ++r)
      Sf[(r + ((lane >> 4) << 3)) * 160 + nt * 16 + (lane & 15)] = c[r];
  }
  __syncthreads();
  if (tid < 16) {
    float mx = -3.4e38f;
    for (int cc = 0; cc < 160; ++cc) mx = fmaxf(mx, Sf[tid * 160 + cc]);
    float sum = 0.f;
    for (int cc = 0; cc < 160; ++cc) {
      float e = __expf(Sf[tid * 160 + cc] - mx);
      Sf[tid * 160 + cc] = e;
      sum += e;
    }
    float inv = 1.f / sum;
    for (int cc = 0; cc < 160; ++cc) Pl[tid * 160 + cc] = f2bf(Sf[tid * 160 + cc] * inv);
  }
  __syncthreads();
  if (wave < 6) {
    int br = wave >> 1, nt = wave & 1;
    v8f c = v8f_zero();
#pragma unroll
    for (int kb = 0; kb < 5; ++kb) {
      us16 pa = *(const us16*)&Pl[(lane & 15) * 160 + kb * 32 + (lane >> 4) * 16];
      us16 tb = *(const us16*)&Tt[(br * 32 + nt * 16 + (lane & 15)) * 160 + kb * 32 + (lane >> 4) * 16];
      c = wmma_bf16(pa, tb, c);
    }
#pragma unroll
    for (int r = 0; r < 8; ++r) {
      int hgl = mt * 16 + r + ((lane >> 4) << 3);
      int j = nt * 16 + (lane & 15);
      acat[((size_t)((br * BW + bw) * Hh + hgl)) * ICn + j] = f2bf(c[r]);
    }
  }
}

// ---------------------------------------------------------------------------
// K4: output conv, operand-swapped: D = Wo_tile(16o x 96) x acat_rows(96 x 16pix)
//     Rows remapped s = h*1280 + bw so lanes -> consecutive w: coalesced
//     residual reads + output writes. 48 WMMA / wave.
// ---------------------------------------------------------------------------
__global__ void k_out(const unsigned short* __restrict__ acat,
                      const unsigned short* __restrict__ wo,
                      const float* __restrict__ so,
                      const float* __restrict__ bo,
                      const float* __restrict__ alphas,
                      const float* __restrict__ feat,
                      float* __restrict__ out) {
  __shared__ __align__(32) unsigned short wLDS[256 * 96];  // 48 KB
  __shared__ float soLDS[256], boLDS[256];
  int tid = threadIdx.x, lane = tid & 31, wave = tid >> 5;
  int sbase = blockIdx.x * 128;          // rows over s = h*1280 + bw
#pragma unroll
  for (int i = 0; i < 12; ++i) {
    int c = tid + i * 256;
    async_load_b128(lds_off(&wLDS[c * 8]), (unsigned)(c * 16),
                    (unsigned long long)(size_t)wo);
  }
  soLDS[tid] = so[tid];
  boLDS[tid] = bo[tid];
  wait_async();
  __syncthreads();
  v8f acc[16];
#pragma unroll
  for (int nt = 0; nt < 16; ++nt) acc[nt] = v8f_zero();
  int s = sbase + wave * 16 + (lane & 15);
  int bwv = s % BW, hv = s / BW;
  const unsigned short* arow =
      acat + (size_t)(bwv * Hh + hv) * 96 + (lane >> 4) * 16;
#pragma unroll
  for (int kb = 0; kb < 3; ++kb) {
    us16 bfrag = *(const us16*)(arow + kb * 32);
#pragma unroll
    for (int nt = 0; nt < 16; ++nt) {
      us16 afrag = *(const us16*)&wLDS[(nt * 16 + (lane & 15)) * 96 + kb * 32 + (lane >> 4) * 16];
      acc[nt] = wmma_bf16(afrag, bfrag, acc[nt]);
    }
  }
  float al = alphas[3];
  int b2 = bwv / Ww, w2 = bwv % Ww;
  size_t pixbase = (size_t)b2 * Cc * HW + hv * Ww + w2;
#pragma unroll
  for (int nt = 0; nt < 16; ++nt) {
#pragma unroll
    for (int r = 0; r < 8; ++r) {
      int oc = nt * 16 + r + ((lane >> 4) << 3);
      size_t gi = pixbase + (size_t)oc * HW;
      float v = acc[nt][r] * soLDS[oc] + boLDS[oc] + feat[gi];
      v = v >= 0.f ? v : al * v;
      out[gi] = v;
    }
  }
}

// ---------------------------------------------------------------------------
extern "C" void kernel_launch(void* const* d_in, const int* in_sizes, int n_in,
                              void* d_out, int out_size, void* d_ws, size_t ws_size,
                              hipStream_t stream) {
  (void)in_sizes; (void)n_in; (void)out_size; (void)ws_size;
  const float* feat = (const float*)d_in[0];

  char* ws = (char*)d_ws;
  size_t off = 0;
  auto alloc = [&](size_t bytes) -> void* {
    void* p = ws + off;
    off = (off + bytes + 255) & ~(size_t)255;
    return p;
  };

  unsigned short* Xb    = (unsigned short*)alloc((size_t)P * 256 * 2);
  unsigned short* q_s   = (unsigned short*)alloc((size_t)BW * Hh * ICn * 2);
  unsigned short* k_s   = (unsigned short*)alloc((size_t)BW * Hh * ICn * 2);
  unsigned short* val_s = (unsigned short*)alloc((size_t)P * ICn * 2);
  unsigned short* tcat  = (unsigned short*)alloc((size_t)3 * BW * ICn * Hh * 2);
  unsigned short* acat  = (unsigned short*)alloc((size_t)3 * BW * Hh * ICn * 2);
  unsigned short* wqkv  = (unsigned short*)alloc(96 * 256 * 2);
  unsigned short* wd    = (unsigned short*)alloc(3 * 9 * 32 * 32 * 2);
  unsigned short* wo    = (unsigned short*)alloc(256 * 96 * 2);
  float* sqkv   = (float*)alloc(96 * 4);
  float* bqkv   = (float*)alloc(96 * 4);
  float* sd     = (float*)alloc(96 * 4);
  float* bd     = (float*)alloc(96 * 4);
  float* so     = (float*)alloc(256 * 4);
  float* bo     = (float*)alloc(256 * 4);
  float* alphas = (float*)alloc(4 * 4);

  PrepIn pin;
  pin.qw = (const float*)d_in[1];  pin.qg = (const float*)d_in[2];
  pin.qb = (const float*)d_in[3];  pin.qm = (const float*)d_in[4];
  pin.qv = (const float*)d_in[5];  pin.qa = (const float*)d_in[6];
  pin.kw = (const float*)d_in[7];  pin.kg = (const float*)d_in[8];
  pin.kbt = (const float*)d_in[9]; pin.km = (const float*)d_in[10];
  pin.kv = (const float*)d_in[11]; pin.ka = (const float*)d_in[12];
  pin.vw = (const float*)d_in[13]; pin.vg = (const float*)d_in[14];
  pin.vb = (const float*)d_in[15]; pin.vm = (const float*)d_in[16];
  pin.vv = (const float*)d_in[17]; pin.va = (const float*)d_in[18];
  pin.d1w = (const float*)d_in[19];
  pin.dg[0] = (const float*)d_in[20]; pin.db[0] = (const float*)d_in[21];
  pin.dm[0] = (const float*)d_in[22]; pin.dv[0] = (const float*)d_in[23];
  pin.d2w = (const float*)d_in[24];
  pin.dg[1] = (const float*)d_in[25]; pin.db[1] = (const float*)d_in[26];
  pin.dm[1] = (const float*)d_in[27]; pin.dv[1] = (const float*)d_in[28];
  pin.d3w = (const float*)d_in[29];
  pin.dg[2] = (const float*)d_in[30]; pin.db[2] = (const float*)d_in[31];
  pin.dm[2] = (const float*)d_in[32]; pin.dv[2] = (const float*)d_in[33];
  pin.ow = (const float*)d_in[34]; pin.og = (const float*)d_in[35];
  pin.ob = (const float*)d_in[36]; pin.om = (const float*)d_in[37];
  pin.ov = (const float*)d_in[38]; pin.oa = (const float*)d_in[39];

  PrepOut pout;
  pout.wqkv = wqkv; pout.wd = wd; pout.wo = wo;
  pout.sqkv = sqkv; pout.bqkv = bqkv; pout.sd = sd; pout.bd = bd;
  pout.so = so; pout.bo = bo; pout.alphas = alphas;

  k_transpose<<<P / 64, 256, 0, stream>>>(feat, Xb);
  k_prep<<<1, 256, 0, stream>>>(pin, pout);
  k_qkv<<<P / 128, 256, 0, stream>>>(Xb, wqkv, sqkv, bqkv, alphas, q_s, k_s, val_s);
  k_dconv<<<dim3(P / 128, 3), 256, 0, stream>>>(val_s, wd, sd, bd, tcat);
  k_attn<<<dim3(Hh / 16, BW), 256, 0, stream>>>(q_s, k_s, tcat, acat);
  k_out<<<P / 128, 256, 0, stream>>>(acat, wo, so, bo, alphas, feat, (float*)d_out);
}
